// Attention_34505767256257
// MI455X (gfx1250) — matrix operand order repaired and checked by reading, not compiled, not run on hardware
//
#include <hip/hip_runtime.h>
#include <hip/hip_bf16.h>

typedef __bf16 bf16;
typedef __attribute__((ext_vector_type(16))) __bf16 v16bf;
typedef __attribute__((ext_vector_type(8)))  float  v8f;

#define NROWS 8192
#define DIN   1024
#define DQKV  3072

union F16x16 { v16bf v; uint4 q[2]; };
union F16x8  { uint4 q; bf16 h[8]; };

// A fragment, 16x32 bf16, from row-major LDS tile (ld elements per row).
// ISA layout: lane m = lane&15, kbase = (lane>>4)*8; elems 0-7 -> k=kb..kb+7,
// elems 8-15 -> k=16+kb..16+kb+7.  Two b128 LDS loads.
__device__ __forceinline__ v16bf frag_a(const bf16* base, int ld, int lane) {
  int m  = lane & 15;
  int kb = (lane >> 4) << 3;
  const bf16* p = base + m * ld + kb;
  F16x16 f;
  f.q[0] = *(const uint4*)(p);
  f.q[1] = *(const uint4*)(p + 16);
  return f.v;
}

// B fragment, 32x16 bf16, from an n-major LDS tile Bt[n][k] (ld = k stride).
// ISA layout: lane n = lane&15, kbase = (lane>>4)*8; elems 0-7 -> k=kb..kb+7,
// elems 8-15 -> k=16+kb..16+kb+7.  Two b128 LDS loads.
__device__ __forceinline__ v16bf frag_b(const bf16* base, int ld, int lane) {
  int n  = lane & 15;
  int kb = (lane >> 4) << 3;
  const bf16* p = base + n * ld + kb;
  F16x16 f;
  f.q[0] = *(const uint4*)(p);
  f.q[1] = *(const uint4*)(p + 16);
  return f.v;
}

__device__ __forceinline__ v8f wmma_bf16(v16bf a, v16bf b, v8f c) {
  return __builtin_amdgcn_wmma_f32_16x16x32_bf16(false, a, false, b, (short)0, c,
                                                 false, false);
}

// ---- CDNA5 async global->LDS DMA (GLOBAL_LOAD_ASYNC_TO_LDS_B128, ASYNCcnt) --
// ISA 10.x: copies global VADDR+offset to LDS VDST+offset, byte for byte;
// the same instruction offset shifts both sides, so one reg pair stages 32B.
__device__ __forceinline__ void async_copy16(unsigned lds_off, const void* g) {
  asm volatile("global_load_async_to_lds_b128 %0, %1, off"
               :: "v"(lds_off), "v"(g) : "memory");
}
__device__ __forceinline__ void async_copy32(unsigned lds_off, const void* g) {
  asm volatile("global_load_async_to_lds_b128 %0, %1, off\n\t"
               "global_load_async_to_lds_b128 %0, %1, off offset:16"
               :: "v"(lds_off), "v"(g) : "memory");
}
__device__ __forceinline__ unsigned lds_addr(const void* p) {
  return (unsigned)(uintptr_t)p;   // generic LDS addr low 32 bits = LDS offset
}
__device__ __forceinline__ void async_wait0() {
  asm volatile("s_wait_asynccnt 0" ::: "memory");
}
// Leave the 4 most recently issued async ops in flight (next tile's DMA).
__device__ __forceinline__ void async_wait4() {
  asm volatile("s_wait_asynccnt 4" ::: "memory");
}

// Stage one 64x32 bf16 Q tile + one 64x32 bf16 K tile via async DMA.
// 128 threads: thread t -> row r = t>>1, half h = (t&1)*16 (32B per side).
__device__ __forceinline__ void stage_qk_async(bf16* Qt, bf16* Kt,
                                               const bf16* qsrc, const bf16* ksrc,
                                               int tid) {
  int r = tid >> 1, h = (tid & 1) * 16;
  async_copy32(lds_addr(Qt + r * 32 + h), qsrc + (size_t)r * DIN + h);
  async_copy32(lds_addr(Kt + r * 32 + h), ksrc + (size_t)r * DIN + h);
}

// ---------------------------------------------------------------------------
// Kernel 1: qkv = x @ Wqkv  (f32 in, bf16 out; q pre-scaled by 1/32)
// BM=64, BN=128, BK=32; 256 threads = 8 waves (4 m-waves x 2 n-waves).
// f32 sources must pass through VGPRs for bf16 conversion -> regular staging.
// ---------------------------------------------------------------------------
__global__ __launch_bounds__(256) void qkv_gemm(const float* __restrict__ x,
                                                const float* __restrict__ Wqkv,
                                                bf16* __restrict__ qb,
                                                bf16* __restrict__ kbuf,
                                                bf16* __restrict__ vb) {
  extern __shared__ char smem[];
  bf16* At = (bf16*)smem;            // [64][32] row-major (k contiguous)
  bf16* Bt = (bf16*)(smem + 4096);   // [128][32] n-major (k contiguous)
  const int tid = threadIdx.x, lane = tid & 31, w = tid >> 5;
  const int wm = w >> 1, wn = w & 1;
  const int row0 = blockIdx.x * 64;
  const int col0 = blockIdx.y * 128;

  v8f c[4] = {};
  for (int kt = 0; kt < DIN / 32; ++kt) {
    const int k0 = kt * 32;
    {  // stage A: 64x32 f32 -> bf16, 8 elems/thread
      int r = tid >> 2, cc = (tid & 3) * 8;
      const float* g = x + (size_t)(row0 + r) * DIN + k0 + cc;
      float4 f0 = *(const float4*)g;
      float4 f1 = *(const float4*)(g + 4);
      F16x8 h;
      h.h[0] = (bf16)f0.x; h.h[1] = (bf16)f0.y; h.h[2] = (bf16)f0.z; h.h[3] = (bf16)f0.w;
      h.h[4] = (bf16)f1.x; h.h[5] = (bf16)f1.y; h.h[6] = (bf16)f1.z; h.h[7] = (bf16)f1.w;
      *(uint4*)(At + r * 32 + cc) = h.q;
    }
    {  // stage B transposed to n-major: 32x128 f32, 16 elems/thread
      int kk = tid >> 5;            // 0..7
      int n4 = (tid & 31) * 4;      // 0..124
      for (int i = 0; i < 4; ++i) {
        int k = kk + i * 8;
        float4 f = *(const float4*)(Wqkv + (size_t)(k0 + k) * DQKV + col0 + n4);
        Bt[(n4 + 0) * 32 + k] = (bf16)f.x;
        Bt[(n4 + 1) * 32 + k] = (bf16)f.y;
        Bt[(n4 + 2) * 32 + k] = (bf16)f.z;
        Bt[(n4 + 3) * 32 + k] = (bf16)f.w;
      }
    }
    __syncthreads();
    v16bf a = frag_a(At + wm * 16 * 32, 32, lane);
#pragma unroll
    for (int nt = 0; nt < 4; ++nt) {
      v16bf b = frag_b(Bt + (wn * 64 + nt * 16) * 32, 32, lane);
      c[nt] = wmma_bf16(a, b, c[nt]);
    }
    __syncthreads();
  }
  const int nl = lane & 15, m0 = (lane >> 4) * 8;
  const int rbase = row0 + wm * 16 + m0;
#pragma unroll
  for (int nt = 0; nt < 4; ++nt) {
    int cg = col0 + wn * 64 + nt * 16 + nl;
    bf16* dst;
    float sc = 1.0f;
    if (cg < 1024)      { dst = qb   + cg;          sc = 0.03125f; }  // fold scale
    else if (cg < 2048) { dst = kbuf + (cg - 1024); }
    else                { dst = vb   + (cg - 2048); }
#pragma unroll
    for (int r = 0; r < 8; ++r)
      dst[(size_t)(rbase + r) * DIN] = (bf16)(c[nt][r] * sc);
  }
}

// ---------------------------------------------------------------------------
// Kernel 2: flash pass 1 — per-row max & sum(exp) of causal-masked S = q k^T.
// 128 threads = 4 waves, each wave owns 16 rows of a 64-row block.
// Q/K tiles double-buffered, staged with async DMA, pipelined against WMMA.
// ---------------------------------------------------------------------------
__global__ __launch_bounds__(128) void attn_stats(const bf16* __restrict__ qb,
                                                  const bf16* __restrict__ kbuf,
                                                  float* __restrict__ rmax,
                                                  float* __restrict__ rsum) {
  extern __shared__ char smem[];
  bf16* Qt[2] = { (bf16*)smem,          (bf16*)(smem + 8192)  };
  bf16* Kt[2] = { (bf16*)(smem + 4096), (bf16*)(smem + 12288) };
  const int tid = threadIdx.x, lane = tid & 31, wv = tid >> 5;
  const int blk = blockIdx.x, r0 = blk * 64;
  const int nl = lane & 15, m0 = (lane >> 4) * 8;

  float m[8], l[8];
#pragma unroll
  for (int r = 0; r < 8; ++r) { m[r] = -3.0e38f; l[r] = 0.f; }

  for (int jb = 0; jb <= blk; ++jb) {
    const int j0 = jb * 64;
    if (jb < blk) __builtin_prefetch(kbuf + (size_t)(j0 + 64) * DIN, 0, 0);
    v8f s[4] = {};
    const bf16* qrow = qb   + (size_t)r0 * DIN;
    const bf16* krow = kbuf + (size_t)j0 * DIN;
    stage_qk_async(Qt[0], Kt[0], qrow, krow, tid);          // prologue: dc=0
    for (int dci = 0; dci < DIN / 32; ++dci) {
      const int cur = dci & 1;
      if (dci + 1 < DIN / 32) {
        stage_qk_async(Qt[1 - cur], Kt[1 - cur],
                       qrow + (dci + 1) * 32, krow + (dci + 1) * 32, tid);
        async_wait4();        // current tile done; next tile's 4 ops in flight
      } else {
        async_wait0();
      }
      __syncthreads();
      v16bf a = frag_a(Qt[cur] + wv * 16 * 32, 32, lane);
#pragma unroll
      for (int nt = 0; nt < 4; ++nt) {
        v16bf b = frag_b(Kt[cur] + nt * 16 * 32, 32, lane); // S[i][j]=sum_d q k
        s[nt] = wmma_bf16(a, b, s[nt]);
      }
      __syncthreads();
    }
#pragma unroll
    for (int r = 0; r < 8; ++r) {
      const int i = r0 + wv * 16 + m0 + r;
      float sv[4], rm = -3.0e38f;
#pragma unroll
      for (int nt = 0; nt < 4; ++nt) {
        int j = j0 + nt * 16 + nl;
        float v = (j <= i) ? s[nt][r] : -3.0e38f;
        sv[nt] = v;
        rm = fmaxf(rm, v);
      }
      rm = fmaxf(rm, __shfl_xor(rm, 1, 32));
      rm = fmaxf(rm, __shfl_xor(rm, 2, 32));
      rm = fmaxf(rm, __shfl_xor(rm, 4, 32));
      rm = fmaxf(rm, __shfl_xor(rm, 8, 32));
      float mn = fmaxf(m[r], rm);
      float ps = 0.f;
#pragma unroll
      for (int nt = 0; nt < 4; ++nt) ps += __expf(sv[nt] - mn);
      ps += __shfl_xor(ps, 1, 32);
      ps += __shfl_xor(ps, 2, 32);
      ps += __shfl_xor(ps, 4, 32);
      ps += __shfl_xor(ps, 8, 32);
      l[r] = l[r] * __expf(m[r] - mn) + ps;
      m[r] = mn;
    }
  }
  if (nl == 0) {
#pragma unroll
    for (int r = 0; r < 8; ++r) {
      int i = r0 + wv * 16 + m0 + r;
      rmax[i] = m[r];
      rsum[i] = l[r];
    }
  }
}

// ---------------------------------------------------------------------------
// Kernel 3: flash pass 2 — O[:, dchunk] = softmax(S) @ V[:, dchunk].
// Grid (128 row-blocks, 4 d-chunks of 256).  S recomputed with async-staged
// WMMA; P re-laid out through LDS as an A fragment; P@V with WMMA.
// ---------------------------------------------------------------------------
__global__ __launch_bounds__(128) void attn_out(const bf16* __restrict__ qb,
                                                const bf16* __restrict__ kbuf,
                                                const bf16* __restrict__ vb,
                                                const float* __restrict__ rmax,
                                                const float* __restrict__ rsum,
                                                bf16* __restrict__ ob) {
  extern __shared__ char smem[];
  bf16* Qt[2] = { (bf16*)smem,          (bf16*)(smem + 8192)  };
  bf16* Kt[2] = { (bf16*)(smem + 4096), (bf16*)(smem + 12288) };
  bf16* Pt = (bf16*)(smem + 16384);   // 4 waves x [16][64]
  bf16* Vt = (bf16*)(smem + 24576);   // [256][64]  (d-major for B frags)
  const int tid = threadIdx.x, lane = tid & 31, wv = tid >> 5;
  const int blk = blockIdx.x, r0 = blk * 64;
  const int dch = blockIdx.y * 256;
  const int nl = lane & 15, m0 = (lane >> 4) * 8;

  float m[8], linv[8];
#pragma unroll
  for (int r = 0; r < 8; ++r) {
    int i = r0 + wv * 16 + m0 + r;
    m[r] = rmax[i];
    linv[r] = 1.0f / rsum[i];
  }
  v8f o[16] = {};
  bf16* Ptw = Pt + wv * 16 * 64;

  for (int jb = 0; jb <= blk; ++jb) {
    const int j0 = jb * 64;
    if (jb < blk) __builtin_prefetch(vb + (size_t)(j0 + 64) * DIN + dch, 0, 0);
    v8f s[4] = {};
    const bf16* qrow = qb   + (size_t)r0 * DIN;
    const bf16* krow = kbuf + (size_t)j0 * DIN;
    stage_qk_async(Qt[0], Kt[0], qrow, krow, tid);
    for (int dci = 0; dci < DIN / 32; ++dci) {
      const int cur = dci & 1;
      if (dci + 1 < DIN / 32) {
        stage_qk_async(Qt[1 - cur], Kt[1 - cur],
                       qrow + (dci + 1) * 32, krow + (dci + 1) * 32, tid);
        async_wait4();
      } else {
        async_wait0();
      }
      __syncthreads();
      v16bf a = frag_a(Qt[cur] + wv * 16 * 32, 32, lane);
#pragma unroll
      for (int nt = 0; nt < 4; ++nt) {
        v16bf b = frag_b(Kt[cur] + nt * 16 * 32, 32, lane);
        s[nt] = wmma_bf16(a, b, s[nt]);
      }
      __syncthreads();
    }
    // P = exp(S - m) (causal-masked), re-layout C->A through LDS
#pragma unroll
    for (int nt = 0; nt < 4; ++nt) {
      int j = j0 + nt * 16 + nl;
#pragma unroll
      for (int r = 0; r < 8; ++r) {
        int i = r0 + wv * 16 + m0 + r;
        float p = (j <= i) ? __expf(s[nt][r] - m[r]) : 0.f;
        Ptw[(m0 + r) * 64 + nt * 16 + nl] = (bf16)p;
      }
    }
    // stage V tile transposed: Vt[d][j], source V[j0..j0+63][dch..dch+255]
    for (int it = 0; it < 16; ++it) {
      int chunk = it * 128 + tid;           // 0..2047 (uint4 chunks)
      int j  = chunk >> 5;                  // 0..63
      int d0 = (chunk & 31) * 8;            // 0..248
      F16x8 hv;
      hv.q = *(const uint4*)(vb + (size_t)(j0 + j) * DIN + dch + d0);
#pragma unroll
      for (int e = 0; e < 8; ++e)
        Vt[(d0 + e) * 64 + j] = hv.h[e];
    }
    __syncthreads();
#pragma unroll
    for (int ks = 0; ks < 2; ++ks) {                  // K = 64 keys, 2 steps
      v16bf a = frag_a(Ptw + ks * 32, 64, lane);
#pragma unroll
      for (int nt = 0; nt < 16; ++nt) {               // 256-wide d chunk
        v16bf b = frag_b(Vt + nt * 16 * 64 + ks * 32, 64, lane);
        o[nt] = wmma_bf16(a, b, o[nt]);
      }
    }
    __syncthreads();
  }
#pragma unroll
  for (int nt = 0; nt < 16; ++nt) {
    int d = dch + nt * 16 + nl;
#pragma unroll
    for (int r = 0; r < 8; ++r) {
      int i = r0 + wv * 16 + m0 + r;
      ob[(size_t)i * DIN + d] = (bf16)(o[nt][r] * linv[r]);
    }
  }
}

// ---------------------------------------------------------------------------
// Kernel 4: out = O @ Wout + bout  (bf16 A async-staged, f32 W converted
// inline, f32 out)
// ---------------------------------------------------------------------------
__global__ __launch_bounds__(256) void out_gemm(const bf16* __restrict__ ob,
                                                const float* __restrict__ Wout,
                                                const float* __restrict__ bout,
                                                float* __restrict__ out) {
  extern __shared__ char smem[];
  bf16* At = (bf16*)smem;            // [64][32]
  bf16* Bt = (bf16*)(smem + 4096);   // [128][32] n-major
  const int tid = threadIdx.x, lane = tid & 31, w = tid >> 5;
  const int wm = w >> 1, wn = w & 1;
  const int row0 = blockIdx.x * 64;
  const int col0 = blockIdx.y * 128;

  v8f c[4] = {};
  for (int kt = 0; kt < DIN / 32; ++kt) {
    const int k0 = kt * 32;
    { int r = tid >> 2, cc = (tid & 3) * 8;     // A: bf16 source, async DMA
      async_copy16(lds_addr(At + r * 32 + cc),
                   ob + (size_t)(row0 + r) * DIN + k0 + cc);
    }
    { int kk = tid >> 5; int n4 = (tid & 31) * 4;
      for (int i = 0; i < 4; ++i) {
        int k = kk + i * 8;
        float4 f = *(const float4*)(Wout + (size_t)(k0 + k) * 1024 + col0 + n4);
        Bt[(n4 + 0) * 32 + k] = (bf16)f.x;
        Bt[(n4 + 1) * 32 + k] = (bf16)f.y;
        Bt[(n4 + 2) * 32 + k] = (bf16)f.z;
        Bt[(n4 + 3) * 32 + k] = (bf16)f.w;
      }
    }
    async_wait0();
    __syncthreads();
    v16bf a = frag_a(At + wm * 16 * 32, 32, lane);
#pragma unroll
    for (int nt = 0; nt < 4; ++nt) {
      v16bf b = frag_b(Bt + (wn * 64 + nt * 16) * 32, 32, lane);
      c[nt] = wmma_bf16(a, b, c[nt]);
    }
    __syncthreads();
  }
  const int nl = lane & 15, m0 = (lane >> 4) * 8;
  const int rbase = row0 + wm * 16 + m0;
#pragma unroll
  for (int nt = 0; nt < 4; ++nt) {
    int cg = col0 + wn * 64 + nt * 16 + nl;
    float bias = bout[cg];
#pragma unroll
    for (int r = 0; r < 8; ++r)
      out[(size_t)(rbase + r) * 1024 + cg] = c[nt][r] + bias;
  }
}

// ---------------------------------------------------------------------------
extern "C" void kernel_launch(void* const* d_in, const int* in_sizes, int n_in,
                              void* d_out, int out_size, void* d_ws, size_t ws_size,
                              hipStream_t stream) {
  (void)in_sizes; (void)n_in; (void)out_size; (void)ws_size;
  const float* x    = (const float*)d_in[0];
  const float* Wqkv = (const float*)d_in[1];
  const float* Wout = (const float*)d_in[2];
  const float* bout = (const float*)d_in[3];
  float* out = (float*)d_out;

  char* ws = (char*)d_ws;
  const size_t sz = (size_t)NROWS * DIN * sizeof(bf16);   // 16 MB each
  bf16* qb = (bf16*)(ws);
  bf16* kb = (bf16*)(ws + sz);
  bf16* vb = (bf16*)(ws + 2 * sz);
  bf16* ob = (bf16*)(ws + 3 * sz);
  float* rmax = (float*)(ws + 4 * sz);
  float* rsum = rmax + NROWS;

  qkv_gemm <<<dim3(NROWS / 64, DQKV / 128), 256, 12 * 1024, stream>>>(x, Wqkv, qb, kb, vb);
  attn_stats<<<dim3(NROWS / 64),            128, 16 * 1024, stream>>>(qb, kb, rmax, rsum);
  attn_out  <<<dim3(NROWS / 64, 4),         128, 56 * 1024, stream>>>(qb, kb, vb, rmax, rsum, ob);
  out_gemm  <<<dim3(NROWS / 64, DIN / 128), 256, 12 * 1024, stream>>>(ob, Wout, bout, out);
}
